// SelfModifyingTitans_72610717106501
// MI455X (gfx1250) — matrix operand order repaired and checked by reading, not compiled, not run on hardware
//
#include <hip/hip_runtime.h>
#include <hip/hip_bf16.h>
#include <math.h>

// ---------------------------------------------------------------------------
// Self-modifying Titans scan for MI455X (gfx1250, wave32, WMMA).
// All matrix contractions use v_wmma_f32_16x16x32_bf16 (M=16 == CHUNK).
// Each wave owns TWO adjacent 16x16 N-tiles (independent accumulators sharing
// the A fragment) and the K loop is unrolled 2x -> 4 WMMAs in flight per wave.
// Weight (B-operand) fragments are gathered from fp32 state with a SINGLE base
// address per fragment + compile-time immediate offsets (row stride fixed at
// 1024 floats), avoiding per-element 64-bit address arithmetic and spills.
// Recurrent state (24 x ~8MB) is kept fp32 in d_ws so it stays resident in the
// 192MB L2 across the 128 serial steps; activations are staged in LDS as bf16.
// NOTE: requires ws_size >= ~270MB (state 201.5MB + z 32MB + out 32MB + temps).
// ---------------------------------------------------------------------------

typedef __attribute__((ext_vector_type(16))) __bf16 v16bf;
typedef __attribute__((ext_vector_type(8)))  float  v8f;

static constexpr int BN = 4;        // batch
static constexpr int TT = 2048;     // sequence
static constexpr int DD = 1024;     // d_model == d_memory (fixed row stride!)
static constexpr int CC = 16;       // chunk
static constexpr int NCHUNK = 128;  // TT / CC
static constexpr int NT = 256;      // threads per block (8 waves)

// per-(batch,module) state slot layout in floats
static constexpr size_t OFF_W1 = 0;
static constexpr size_t OFF_B1 = (size_t)DD * DD;
static constexpr size_t OFF_W2 = OFF_B1 + DD;
static constexpr size_t OFF_B2 = OFF_W2 + (size_t)DD * DD;
static constexpr size_t SLOT   = OFF_B2 + DD;
static constexpr size_t BSTR   = 6 * SLOT;

__device__ __forceinline__ float sigf(float x) { return 1.f / (1.f + __expf(-x)); }

__device__ __forceinline__ v8f wmma_bf(v16bf a, v16bf b, v8f c) {
  return __builtin_amdgcn_wmma_f32_16x16x32_bf16(false, a, false, b, (short)0, c,
                                                 false, false);
}

__device__ __forceinline__ void tile_coords(int& n, int& mb) {
  int lane = threadIdx.x & 31;
  n = lane & 15;
  mb = (lane >> 4) * 8;   // C/D layout: lanes 0-15 rows 0-7, lanes 16-31 rows 8-15
}

// ---- fragment loaders (16-bit A 16x32 layout per ISA 7.12.2) ----
// A: 16 x 1024 bf16 row-major in LDS; fetch the 16x32 slab at column k0.
__device__ __forceinline__ v16bf load_a(const __bf16* A, int k0) {
  const int lane = threadIdx.x & 31;
  const int m = lane & 15;
  const int hi8 = (lane >> 4) ? 8 : 0;
  const __bf16* row = A + m * 1024 + k0;
  v16bf f;
#pragma unroll
  for (int j = 0; j < 16; ++j) {
    int vg = j >> 1, half = j & 1;
    int k = ((vg & 4) ? 16 : 0) + hi8 + ((vg & 3) * 2 + half);
    f[j] = row[k];
  }
  return f;
}
// A^T slab with true K=16 (zero-padded to 32): A_lds holds A[c][m] (16 x 1024).
__device__ __forceinline__ v16bf load_aT16(const __bf16* A, int m0) {
  const int lane = threadIdx.x & 31;
  const int m = m0 + (lane & 15);
  const int hi8 = (lane >> 4) ? 8 : 0;
  v16bf f;
#pragma unroll
  for (int j = 0; j < 16; ++j) {
    int vg = j >> 1, half = j & 1;
    int k = ((vg & 4) ? 16 : 0) + hi8 + ((vg & 3) * 2 + half);
    f[j] = (k < 16) ? A[k * 1024 + m] : (__bf16)0.f;
  }
  return f;
}
// B[k][n] = W[n][k]  (W: N x 1024 fp32 row-major) -> multiply by W^T.
// One base pointer per lane (n = lane%16); table-order k at immediate offsets.
__device__ __forceinline__ v16bf load_bT(const float* W, int n0, int k0) {
  const int lane = threadIdx.x & 31;
  const float* p = W + (((size_t)(n0 + (lane & 15))) << 10) + k0 + ((lane >> 4) << 3);
  v16bf f;
#pragma unroll
  for (int j = 0; j < 16; ++j) f[j] = (__bf16)p[j < 8 ? j : j + 8];
  return f;
}
// B[k][n] = W[k][n]  (W: 1024 x 1024 fp32 row-major). Column n=lane%16, table-order k.
__device__ __forceinline__ v16bf load_bN(const float* W, int n0, int k0) {
  const int lane = threadIdx.x & 31;
  const float* p = W + (((size_t)(k0 + ((lane >> 4) << 3))) << 10) + n0 + (lane & 15);
  v16bf f;
#pragma unroll
  for (int j = 0; j < 16; ++j) f[j] = (__bf16)p[(size_t)(j < 8 ? j : j + 8) << 10];
  return f;
}
// B 16x16 slab zero-padded to K=32 from LDS bf16 [16][1024]: B[c][n].
__device__ __forceinline__ v16bf load_bK16(const __bf16* Bm, int n0) {
  const int lane = threadIdx.x & 31;
  v16bf f;
  const __bf16* p = Bm + ((lane >> 4) << 3) * 1024 + n0 + (lane & 15);
#pragma unroll
  for (int j = 0; j < 16; ++j) f[j] = (j < 8) ? p[j * 1024] : (__bf16)0.f;
  return f;
}

// Dual-tile GEMM: C0/C1 (16x16 each) = A(16xK) @ W^T rows [n0,n0+16)/[n0+16,n0+32).
// Shared A fragment, independent accumulators, 2x K unroll -> 4 WMMAs in flight.
// Each lane prefetches one of the 32 weight rows needed at the next K slab.
__device__ __forceinline__ void gemm_tile_T2(const __bf16* A, const float* W, int n0,
                                             v8f& acc0, v8f& acc1) {
  const int lane = threadIdx.x & 31;
#pragma unroll 2
  for (int k0 = 0; k0 < DD; k0 += 32) {
    __builtin_prefetch(&W[(((size_t)(n0 + lane)) << 10) + ((k0 + 32) & (DD - 1))], 0,
                       1);
    v16bf a = load_a(A, k0);
    acc0 = wmma_bf(a, load_bT(W, n0, k0), acc0);
    acc1 = wmma_bf(a, load_bT(W, n0 + 16, k0), acc1);
  }
}
// Dual-tile variant with W row-major 1024 x 1024 (multiply by W, not W^T).
__device__ __forceinline__ void gemm_tile_N2(const __bf16* A, const float* W, int n0,
                                             v8f& acc0, v8f& acc1) {
#pragma unroll 2
  for (int k0 = 0; k0 < DD; k0 += 32) {
    v16bf a = load_a(A, k0);
    acc0 = wmma_bf(a, load_bN(W, n0, k0), acc0);
    acc1 = wmma_bf(a, load_bN(W, n0 + 16, k0), acc1);
  }
}

__device__ __forceinline__ void stage_bf16(const float* src, __bf16* dst, int n) {
  for (int i = threadIdx.x; i < n; i += NT) dst[i] = (__bf16)src[i];
}

// S(Mrows x 1024) = clamp( a*S - e*(A^T @ B) ), A,B: 16 x 1024 bf16 in LDS (K=16
// rank-16 outer product via zero-padded WMMA). Two passes: norm then scaled write.
// Each wave processes pairs of adjacent N tiles sharing the A^T fragment.
__device__ void rank16_clamp_update(const __bf16* A, const __bf16* Bm, float* S,
                                    int Mrows, float a, float e, float maxn,
                                    float* red) {
  const int wv = threadIdx.x >> 5;
  int n, mb;
  tile_coords(n, mb);
  const int npairs = (Mrows >> 4) * 32;
  float ss = 0.f;
  for (int t = wv; t < npairs; t += 8) {
    int m0 = (t >> 5) << 4, n0 = (t & 31) << 5;
    v16bf at = load_aT16(A, m0);
    v8f g0 = {}, g1 = {};
    g0 = wmma_bf(at, load_bK16(Bm, n0), g0);
    g1 = wmma_bf(at, load_bK16(Bm, n0 + 16), g1);
    const float* Sp = S + (((size_t)(m0 + mb)) << 10) + n0 + n;
#pragma unroll
    for (int r = 0; r < 8; ++r) {
      float t0 = a * Sp[(size_t)r << 10] - e * g0[r];
      float t1 = a * Sp[((size_t)r << 10) + 16] - e * g1[r];
      ss += t0 * t0 + t1 * t1;
    }
  }
  red[threadIdx.x] = ss;
  __syncthreads();
  if (threadIdx.x == 0) {
    float s = 0.f;
    for (int i = 0; i < NT; ++i) s += red[i];
    red[NT] = fminf(maxn / (sqrtf(s) + 1e-8f), 1.f);
  }
  __syncthreads();
  const float sc = red[NT];
  for (int t = wv; t < npairs; t += 8) {
    int m0 = (t >> 5) << 4, n0 = (t & 31) << 5;
    v16bf at = load_aT16(A, m0);
    v8f g0 = {}, g1 = {};
    g0 = wmma_bf(at, load_bK16(Bm, n0), g0);
    g1 = wmma_bf(at, load_bK16(Bm, n0 + 16), g1);
    float* Sp = S + (((size_t)(m0 + mb)) << 10) + n0 + n;
#pragma unroll
    for (int r = 0; r < 8; ++r) {
      size_t i0 = (size_t)r << 10;
      Sp[i0] = (a * Sp[i0] - e * g0[r]) * sc;
      Sp[i0 + 16] = (a * Sp[i0 + 16] - e * g1[r]) * sc;
    }
  }
  __syncthreads();
}

__device__ void clamped_vec_update(float* S, const float* G, int n, float a, float e,
                                   float maxn, float* red) {
  float ss = 0.f;
  for (int i = threadIdx.x; i < n; i += NT) {
    float t = a * S[i] - e * G[i];
    ss += t * t;
  }
  red[threadIdx.x] = ss;
  __syncthreads();
  if (threadIdx.x == 0) {
    float s = 0.f;
    for (int i = 0; i < NT; ++i) s += red[i];
    red[NT] = fminf(maxn / (sqrtf(s) + 1e-8f), 1.f);
  }
  __syncthreads();
  const float sc = red[NT];
  for (int i = threadIdx.x; i < n; i += NT) S[i] = (a * S[i] - e * G[i]) * sc;
  __syncthreads();
}

// ---------------------------------------------------------------------------
// Kernels
// ---------------------------------------------------------------------------

__global__ void bcast_kernel(const float* __restrict__ src, float* __restrict__ dst,
                             size_t n, size_t bstride, int B) {
  size_t i = (size_t)blockIdx.x * blockDim.x + threadIdx.x;
  if (i < n) {
    float v = src[i];
    for (int b = 0; b < B; ++b) dst[b * bstride + i] = v;
  }
}

// C = A @ W^T + bias (all 1024-stride).  grid: (M/16, 1024/256), block 256.
__global__ void gemm_proj(const float* __restrict__ A, const float* __restrict__ W,
                          const float* __restrict__ bias, float* __restrict__ Cm) {
  extern __shared__ char smem[];
  __bf16* Ab = (__bf16*)smem;                       // 16*1024 bf16
  const float* Arow = A + (((size_t)blockIdx.x) << 4 << 10);
  stage_bf16(Arow, Ab, CC * DD);
  __syncthreads();
  const int wv = threadIdx.x >> 5;
  int n, mb;
  tile_coords(n, mb);
  const int n0 = blockIdx.y * 256 + wv * 32;
  v8f a0 = {}, a1 = {};
  gemm_tile_T2(Ab, W, n0, a0, a1);
  const float bv0 = bias[n0 + n];
  const float bv1 = bias[n0 + 16 + n];
  float* Crow = Cm + (((size_t)blockIdx.x) << 4 << 10) + (((size_t)mb) << 10) + n0 + n;
#pragma unroll
  for (int r = 0; r < 8; ++r) {
    Crow[(size_t)r << 10] = a0[r] + bv0;
    Crow[((size_t)r << 10) + 16] = a1[r] + bv1;
  }
}

// Forward of z-fed modules.  grid (5, B): m = 0..4 -> k,v,q,eta,alpha.
__global__ void titans_fwd(const float* __restrict__ z, const float* __restrict__ S,
                           float* __restrict__ kvq, float* __restrict__ scal,
                           int step) {
  extern __shared__ char smem[];
  __bf16* Az = (__bf16*)smem;                  // 32KB  bf16 z chunk
  __bf16* Hb = (__bf16*)(smem + 32768);        // 32KB  bf16 hidden
  float*  Of = (float*)(smem + 65536);         // 64KB  fp32 out
  float*  red = (float*)(smem + 131072);       // 288 floats
  const int m = blockIdx.x;
  const int b = blockIdx.y;
  const int wv = threadIdx.x >> 5;
  const float* zc = z + ((size_t)b * TT + (size_t)step * CC) * DD;
  stage_bf16(zc, Az, CC * DD);
  __syncthreads();
  const float* st = S + (size_t)b * BSTR + (size_t)m * SLOT;
  const float* W1 = st + OFF_W1;
  const float* b1 = st + OFF_B1;
  const float* W2 = st + OFF_W2;
  const float* b2 = st + OFF_B2;
  int n, mb;
  tile_coords(n, mb);
  // stage 1: h = silu(z @ W2^T + b2)
  for (int t = wv; t < 32; t += 8) {
    int n0 = t << 5;
    v8f a0 = {}, a1 = {};
    gemm_tile_T2(Az, W2, n0, a0, a1);
    float bi0 = b2[n0 + n], bi1 = b2[n0 + 16 + n];
    __bf16* Hp = Hb + (((size_t)mb) << 10) + n0 + n;
#pragma unroll
    for (int r = 0; r < 8; ++r) {
      float p0 = a0[r] + bi0, p1 = a1[r] + bi1;
      Hp[(size_t)r << 10] = (__bf16)(p0 * sigf(p0));
      Hp[((size_t)r << 10) + 16] = (__bf16)(p1 * sigf(p1));
    }
  }
  __syncthreads();
  if (m < 3) {  // k, v, q : out = h @ W1^T + b1 + z  (residual)
    for (int t = wv; t < 32; t += 8) {
      int n0 = t << 5;
      v8f a0 = {}, a1 = {};
      gemm_tile_T2(Hb, W1, n0, a0, a1);
      float bi0 = b1[n0 + n], bi1 = b1[n0 + 16 + n];
      size_t base = (((size_t)mb) << 10) + n0 + n;
#pragma unroll
      for (int r = 0; r < 8; ++r) {
        size_t i0 = base + ((size_t)r << 10);
        Of[i0] = a0[r] + bi0 + zc[i0];
        Of[i0 + 16] = a1[r] + bi1 + zc[i0 + 16];
      }
    }
    __syncthreads();
    float* dst = kvq + ((size_t)b * 3 + m) * CC * DD;
    if (m == 1) {  // v: no norm
      for (int i = threadIdx.x; i < CC * DD; i += NT) dst[i] = Of[i];
    } else {       // k, q: per-row l2 norm
      const int row = threadIdx.x & 15, cg = threadIdx.x >> 4;
      float ssq = 0.f;
      for (int c2 = cg; c2 < DD; c2 += 16) {
        float x = Of[row * 1024 + c2];
        ssq += x * x;
      }
      red[row * 16 + cg] = ssq;
      __syncthreads();
      if (cg == 0) {
        float s = 0.f;
        for (int g2 = 0; g2 < 16; ++g2) s += red[row * 16 + g2];
        red[256 + row] = 1.f / fmaxf(sqrtf(s), 1e-12f);
      }
      __syncthreads();
      float inv = red[256 + row];
      for (int c2 = cg; c2 < DD; c2 += 16)
        dst[row * 1024 + c2] = Of[row * 1024 + c2] * inv;
    }
  } else {  // eta (m=3) / alpha (m=4): scalar head, sigmoid, mean over chunk
    const int row = threadIdx.x & 15, cg = threadIdx.x >> 4;
    float p = 0.f;
    for (int c2 = cg; c2 < DD; c2 += 16) p += (float)Hb[row * 1024 + c2] * W1[c2];
    red[row * 16 + cg] = p;
    __syncthreads();
    if (threadIdx.x < 16) {
      float s = b1[0];
      for (int g2 = 0; g2 < 16; ++g2) s += red[threadIdx.x * 16 + g2];
      red[256 + threadIdx.x] = sigf(s);
    }
    __syncthreads();
    if (threadIdx.x == 0) {
      float mn = 0.f;
      for (int c2 = 0; c2 < 16; ++c2) mn += red[256 + c2];
      scal[b * 2 + (m - 3)] = mn * (1.f / 16.f);  // [0]=eta_mean, [1]=alpha_mean
    }
  }
}

// v_hat for all 6 modules from v, plus o = mlp(q, memory).  grid (7, B).
__global__ void titans_vhat(const float* __restrict__ kvq, const float* __restrict__ S,
                            float* __restrict__ vhat, float* __restrict__ outbuf,
                            int step) {
  extern __shared__ char smem[];
  __bf16* Ain = (__bf16*)smem;                 // 32KB
  __bf16* Hb = (__bf16*)(smem + 32768);        // 32KB
  float*  red = (float*)(smem + 65536);        // 320 floats
  const int DOv[7] = {1024, 1024, 1024, 1, 1, 1024, 1024};
  const int RESv[7] = {1, 1, 1, 0, 0, 1, 1};
  const int j = blockIdx.x;  // 0..5: vhat_j from v ; 6: o from q via memory module
  const int b = blockIdx.y;
  const int mm = (j == 6) ? 5 : j;
  const int wv = threadIdx.x >> 5;
  const float* in = kvq + ((size_t)b * 3 + ((j == 6) ? 2 : 1)) * CC * DD;
  stage_bf16(in, Ain, CC * DD);
  __syncthreads();
  const float* st = S + (size_t)b * BSTR + (size_t)mm * SLOT;
  const float* W1 = st + OFF_W1;
  const float* b1 = st + OFF_B1;
  const float* W2 = st + OFF_W2;
  const float* b2 = st + OFF_B2;
  int n, mb;
  tile_coords(n, mb);
  for (int t = wv; t < 32; t += 8) {
    int n0 = t << 5;
    v8f a0 = {}, a1 = {};
    gemm_tile_T2(Ain, W2, n0, a0, a1);
    float bi0 = b2[n0 + n], bi1 = b2[n0 + 16 + n];
    __bf16* Hp = Hb + (((size_t)mb) << 10) + n0 + n;
#pragma unroll
    for (int r = 0; r < 8; ++r) {
      float p0 = a0[r] + bi0, p1 = a1[r] + bi1;
      Hp[(size_t)r << 10] = (__bf16)(p0 * sigf(p0));
      Hp[((size_t)r << 10) + 16] = (__bf16)(p1 * sigf(p1));
    }
  }
  __syncthreads();
  float* dst = (j == 6) ? outbuf + ((size_t)b * TT + (size_t)step * CC) * DD
                        : vhat + ((size_t)b * 6 + j) * CC * DD;
  if (DOv[j] == 1024) {
    for (int t = wv; t < 32; t += 8) {
      int n0 = t << 5;
      v8f a0 = {}, a1 = {};
      gemm_tile_T2(Hb, W1, n0, a0, a1);
      float bi0 = b1[n0 + n], bi1 = b1[n0 + 16 + n];
      size_t base = (((size_t)mb) << 10) + n0 + n;
#pragma unroll
      for (int r = 0; r < 8; ++r) {
        size_t i0 = base + ((size_t)r << 10);
        dst[i0] = a0[r] + bi0 + (RESv[j] ? in[i0] : 0.f);
        dst[i0 + 16] = a1[r] + bi1 + (RESv[j] ? in[i0 + 16] : 0.f);
      }
    }
  } else {  // eta/alpha heads: (16x1), no residual
    const int row = threadIdx.x & 15, cg = threadIdx.x >> 4;
    float p = 0.f;
    for (int c2 = cg; c2 < DD; c2 += 16) p += (float)Hb[row * 1024 + c2] * W1[c2];
    red[row * 16 + cg] = p;
    __syncthreads();
    if (threadIdx.x < 16) {
      float s = b1[0];
      for (int g2 = 0; g2 < 16; ++g2) s += red[threadIdx.x * 16 + g2];
      dst[threadIdx.x * 1024] = s;
    }
  }
}

// DGD grads + clamped in-place state update.  grid (6, B), one block owns (b,m).
__global__ void titans_grad(const float* __restrict__ kvq,
                            const float* __restrict__ vhat, float* __restrict__ S,
                            const float* __restrict__ scal) {
  extern __shared__ char smem[];
  __bf16* Kb = (__bf16*)smem;                  //   0..32KB  bf16 k (input x)
  __bf16* Hb = (__bf16*)(smem + 32768);        //  32..64KB  bf16 h
  float*  Pre = (float*)(smem + 65536);        //  64..128KB fp32 pre-activation
  __bf16* Eb = (__bf16*)(smem + 131072);       // 128..160KB bf16 error
  __bf16* Db = (__bf16*)(smem + 163840);       // 160..192KB bf16 dl_dpre
  float*  g1 = (float*)(smem + 196608);        // db1 grads / errs (do=1)
  float*  g2 = (float*)(smem + 200704);        // db2 grads
  float*  red = (float*)(smem + 204800);       // 320 floats
  const int DOv[6] = {1024, 1024, 1024, 1, 1, 1024};
  const int RESv[6] = {1, 1, 1, 0, 0, 1};
  const int m = blockIdx.x, b = blockIdx.y;
  const int wv = threadIdx.x >> 5;
  const int do1 = DOv[m];
  const float* kin = kvq + ((size_t)b * 3) * CC * DD;              // x = k (normed)
  const float* vh = vhat + ((size_t)b * 6 + m) * CC * DD;          // target
  float* st = S + (size_t)b * BSTR + (size_t)m * SLOT;
  float* W1 = st + OFF_W1;
  float* b1 = st + OFF_B1;
  float* W2 = st + OFF_W2;
  float* b2 = st + OFF_B2;
  const float e_ = scal[b * 2 + 0];  // eta mean  -> scales grad
  const float a_ = scal[b * 2 + 1];  // alpha mean -> scales old state
  stage_bf16(kin, Kb, CC * DD);
  for (int i = threadIdx.x; i < 1024; i += NT) {
    g1[i] = 0.f;
    g2[i] = 0.f;
  }
  __syncthreads();
  int n, mb;
  tile_coords(n, mb);
  // pre = k @ W2^T + b2 ; h = silu(pre)
  for (int t = wv; t < 32; t += 8) {
    int n0 = t << 5;
    v8f a0 = {}, a1 = {};
    gemm_tile_T2(Kb, W2, n0, a0, a1);
    float bi0 = b2[n0 + n], bi1 = b2[n0 + 16 + n];
    size_t base = (((size_t)mb) << 10) + n0 + n;
#pragma unroll
    for (int r = 0; r < 8; ++r) {
      size_t i0 = base + ((size_t)r << 10);
      float p0 = a0[r] + bi0, p1 = a1[r] + bi1;
      Pre[i0] = p0;
      Pre[i0 + 16] = p1;
      Hb[i0] = (__bf16)(p0 * sigf(p0));
      Hb[i0 + 16] = (__bf16)(p1 * sigf(p1));
    }
  }
  __syncthreads();
  // error = (pred - v_hat)/C ; db1 = colsum(error)
  if (do1 == 1024) {
    for (int t = wv; t < 32; t += 8) {
      int n0 = t << 5;
      v8f a0 = {}, a1 = {};
      gemm_tile_T2(Hb, W1, n0, a0, a1);
      float bi0 = b1[n0 + n], bi1 = b1[n0 + 16 + n];
      float cs0 = 0.f, cs1 = 0.f;
      size_t base = (((size_t)mb) << 10) + n0 + n;
#pragma unroll
      for (int r = 0; r < 8; ++r) {
        size_t i0 = base + ((size_t)r << 10);
        float pr0 = a0[r] + bi0 + (RESv[m] ? kin[i0] : 0.f);
        float pr1 = a1[r] + bi1 + (RESv[m] ? kin[i0 + 16] : 0.f);
        float er0 = (pr0 - vh[i0]) * (1.f / 16.f);
        float er1 = (pr1 - vh[i0 + 16]) * (1.f / 16.f);
        Eb[i0] = (__bf16)er0;
        Eb[i0 + 16] = (__bf16)er1;
        cs0 += er0;
        cs1 += er1;
      }
      atomicAdd(&g1[n0 + n], cs0);
      atomicAdd(&g1[n0 + 16 + n], cs1);
    }
  } else {
    const int row = threadIdx.x & 15, cg = threadIdx.x >> 4;
    float p = 0.f;
    for (int c2 = cg; c2 < DD; c2 += 16) p += (float)Hb[row * 1024 + c2] * W1[c2];
    red[row * 16 + cg] = p;
    __syncthreads();
    if (threadIdx.x < 16) {
      float s = b1[0];
      for (int q2 = 0; q2 < 16; ++q2) s += red[threadIdx.x * 16 + q2];
      g1[threadIdx.x] = (s - vh[threadIdx.x * 1024]) * (1.f / 16.f);
    }
  }
  __syncthreads();
  // dl_dpre = (error @ W1) * silu'(pre) ; db2 = colsum(dl_dpre)
  if (do1 == 1024) {
    for (int t = wv; t < 32; t += 8) {
      int n0 = t << 5;
      v8f a0 = {}, a1 = {};
      gemm_tile_N2(Eb, W1, n0, a0, a1);
      float cs0 = 0.f, cs1 = 0.f;
      size_t base = (((size_t)mb) << 10) + n0 + n;
#pragma unroll
      for (int r = 0; r < 8; ++r) {
        size_t i0 = base + ((size_t)r << 10);
        float p0 = Pre[i0], s0 = sigf(p0);
        float p1 = Pre[i0 + 16], s1 = sigf(p1);
        float dp0 = a0[r] * (s0 * (1.f + p0 * (1.f - s0)));
        float dp1 = a1[r] * (s1 * (1.f + p1 * (1.f - s1)));
        Db[i0] = (__bf16)dp0;
        Db[i0 + 16] = (__bf16)dp1;
        cs0 += dp0;
        cs1 += dp1;
      }
      atomicAdd(&g2[n0 + n], cs0);
      atomicAdd(&g2[n0 + 16 + n], cs1);
    }
  } else {
    for (int i = threadIdx.x; i < CC * DD; i += NT) {
      int c2 = i >> 10, hh = i & 1023;
      float p = Pre[i], sg = sigf(p);
      float dp = g1[c2] * W1[hh] * (sg * (1.f + p * (1.f - sg)));
      Db[i] = (__bf16)dp;
      atomicAdd(&g2[hh], dp);
    }
  }
  __syncthreads();
  // ---- clamped updates (norm clamp: sqrt(numel) per-batch-tensor) ----
  if (do1 == 1024) {
    rank16_clamp_update(Eb, Hb, W1, 1024, a_, e_, 1024.f, red);  // dW1 = err^T h
    clamped_vec_update(b1, g1, 1024, a_, e_, 32.f, red);
  } else {
    // W1 row (1 x 1024): g[h] = sum_c err[c] * h[c][h]
    float ss = 0.f;
    for (int h = threadIdx.x; h < DD; h += NT) {
      float g = 0.f;
#pragma unroll
      for (int c2 = 0; c2 < 16; ++c2) g += g1[c2] * (float)Hb[c2 * 1024 + h];
      float t = a_ * W1[h] - e_ * g;
      ss += t * t;
    }
    red[threadIdx.x] = ss;
    __syncthreads();
    if (threadIdx.x == 0) {
      float s = 0.f;
      for (int i = 0; i < NT; ++i) s += red[i];
      red[NT] = fminf(32.f / (sqrtf(s) + 1e-8f), 1.f);
    }
    __syncthreads();
    float sc = red[NT];
    for (int h = threadIdx.x; h < DD; h += NT) {
      float g = 0.f;
#pragma unroll
      for (int c2 = 0; c2 < 16; ++c2) g += g1[c2] * (float)Hb[c2 * 1024 + h];
      W1[h] = (a_ * W1[h] - e_ * g) * sc;
    }
    __syncthreads();
    if (threadIdx.x == 0) {  // b1 scalar
      float gs = 0.f;
#pragma unroll
      for (int c2 = 0; c2 < 16; ++c2) gs += g1[c2];
      float t = a_ * b1[0] - e_ * gs;
      b1[0] = t * fminf(1.f / (fabsf(t) + 1e-8f), 1.f);
    }
    __syncthreads();
  }
  rank16_clamp_update(Db, Kb, W2, 1024, a_, e_, 1024.f, red);    // dW2 = dp^T k
  clamped_vec_update(b2, g2, 1024, a_, e_, 32.f, red);
}

__global__ void ln_kernel(const float* __restrict__ X, const float* __restrict__ g,
                          const float* __restrict__ be, float* __restrict__ Y) {
  __shared__ float red[NT + 2];
  const size_t row = blockIdx.x;
  const float* x = X + row * (size_t)DD;
  float* y = Y + row * (size_t)DD;
  float s = 0.f;
  for (int i = threadIdx.x; i < DD; i += NT) s += x[i];
  red[threadIdx.x] = s;
  __syncthreads();
  if (threadIdx.x == 0) {
    float t = 0.f;
    for (int i = 0; i < NT; ++i) t += red[i];
    red[NT] = t * (1.f / DD);
  }
  __syncthreads();
  const float mu = red[NT];
  float v = 0.f;
  for (int i = threadIdx.x; i < DD; i += NT) {
    float d = x[i] - mu;
    v += d * d;
  }
  __syncthreads();
  red[threadIdx.x] = v;
  __syncthreads();
  if (threadIdx.x == 0) {
    float t = 0.f;
    for (int i = 0; i < NT; ++i) t += red[i];
    red[NT + 1] = rsqrtf(t * (1.f / DD) + 1e-5f);
  }
  __syncthreads();
  const float inv = red[NT + 1];
  for (int i = threadIdx.x; i < DD; i += NT)
    y[i] = (x[i] - mu) * inv * g[i] + be[i];
}

// ---------------------------------------------------------------------------

static constexpr int GEMM_SHM = 32768;
static constexpr int FWD_SHM = 131072 + 288 * 4;
static constexpr int VHAT_SHM = 65536 + 320 * 4;
static constexpr int GRAD_SHM = 204800 + 320 * 4;

extern "C" void kernel_launch(void* const* d_in, const int* in_sizes, int n_in,
                              void* d_out, int out_size, void* d_ws, size_t ws_size,
                              hipStream_t stream) {
  const float* x = (const float*)d_in[0];
  const float* in_W = (const float*)d_in[25];
  const float* in_b = (const float*)d_in[26];
  const float* out_W = (const float*)d_in[27];
  const float* out_b = (const float*)d_in[28];
  const float* ln_g = (const float*)d_in[29];
  const float* ln_b = (const float*)d_in[30];

  float* S = (float*)d_ws;                                    // 24 * SLOT floats
  float* z = S + (size_t)BN * BSTR;                           // (B,T,D)
  float* outbuf = z + (size_t)BN * TT * DD;                   // (B,T,D)
  float* kvq = outbuf + (size_t)BN * TT * DD;                 // (B,3,16,D)
  float* vhat = kvq + (size_t)BN * 3 * CC * DD;               // (B,6,16,D)
  float* scal = vhat + (size_t)BN * 6 * CC * DD;              // (B,2)
  float* normed = z;                                          // reuse z after scan

  (void)hipFuncSetAttribute(reinterpret_cast<const void*>(titans_fwd),
                            hipFuncAttributeMaxDynamicSharedMemorySize, FWD_SHM);
  (void)hipFuncSetAttribute(reinterpret_cast<const void*>(titans_vhat),
                            hipFuncAttributeMaxDynamicSharedMemorySize, VHAT_SHM);
  (void)hipFuncSetAttribute(reinterpret_cast<const void*>(titans_grad),
                            hipFuncAttributeMaxDynamicSharedMemorySize, GRAD_SHM);
  (void)hipFuncSetAttribute(reinterpret_cast<const void*>(gemm_proj),
                            hipFuncAttributeMaxDynamicSharedMemorySize, GEMM_SHM);

  // broadcast initial module params into the per-batch state slots
  for (int m = 0; m < 6; ++m) {
    for (int p = 0; p < 4; ++p) {
      int idx = 1 + m * 4 + p;
      size_t off = (p == 0) ? OFF_W1 : (p == 1) ? OFF_B1 : (p == 2) ? OFF_W2 : OFF_B2;
      size_t nel = (size_t)in_sizes[idx];
      bcast_kernel<<<(unsigned)((nel + 255) / 256), 256, 0, stream>>>(
          (const float*)d_in[idx], S + (size_t)m * SLOT + off, nel, BSTR, BN);
    }
  }

  // z = x @ in_W^T + in_b
  gemm_proj<<<dim3(BN * TT / 16, DD / 256), NT, GEMM_SHM, stream>>>(x, in_W, in_b, z);
  // 128-step serial scan
  for (int s = 0; s < NCHUNK; ++s) {
    titans_fwd<<<dim3(5, BN), NT, FWD_SHM, stream>>>(z, S, kvq, scal, s);
    titans_vhat<<<dim3(7, BN), NT, VHAT_SHM, stream>>>(kvq, S, vhat, outbuf, s);
    titans_grad<<<dim3(6, BN), NT, GRAD_SHM, stream>>>(kvq, vhat, S, scal);
  }
  // LayerNorm + output projection
  ln_kernel<<<BN * TT, NT, 0, stream>>>(outbuf, ln_g, ln_b, normed);
  gemm_proj<<<dim3(BN * TT / 16, DD / 256), NT, GEMM_SHM, stream>>>(
      normed, out_W, out_b, (float*)d_out);
}
